// VQAudioQuantizer_11922829214091
// MI455X (gfx1250) — matrix operand order repaired and checked by reading, not compiled, not run on hardware
//
#include <hip/hip_runtime.h>
#include <hip/hip_bf16.h>

// ---------------------------------------------------------------------------
// VQ quantizer for gfx1250: TDM-staged bf16 WMMA distance GEMM + fused argmin.
// dims hardcoded from reference: B=4, T=4096, D=256, K=8192
// ---------------------------------------------------------------------------

typedef __bf16 bf16_t;
typedef __attribute__((ext_vector_type(16))) __bf16 v16bf;
typedef __attribute__((ext_vector_type(8)))  __bf16 v8bf;
typedef __attribute__((ext_vector_type(8)))  float  v8f;
typedef __attribute__((ext_vector_type(4)))  unsigned int u32x4;
typedef __attribute__((ext_vector_type(8)))  int i32x8;
typedef __attribute__((ext_vector_type(4)))  int i32x4;

constexpr int Bc = 4;
constexpr int Tc = 4096;
constexpr int Dc = 256;
constexpr int Kc = 8192;
constexpr int Mc = Bc * Tc;          // 16384 rows of z
constexpr float COMMIT_W = 1.0f;

constexpr int TILE_N   = 64;                  // codes per TDM tile
constexpr int NTILES   = Kc / TILE_N;         // 128 tiles
constexpr int ROW_B    = Dc * 2;              // 512 bytes per codebook row (bf16)
constexpr int PITCH_B  = ROW_B + 16;          // 528: TDM pad -> conflict-free banks
constexpr int TILE_LDS = TILE_N * PITCH_B;    // 33792 bytes per buffer

// ---------------------------------------------------------------------------
// 1) z -> bf16(-2*z)   (pre-scale so WMMA accumulates -2 * z . e directly)
// ---------------------------------------------------------------------------
__global__ void prep_z_kernel(const float* __restrict__ z, bf16_t* __restrict__ zb) {
    int i = blockIdx.x * 256 + threadIdx.x;   // grid covers Mc*Dc exactly
    zb[i] = (bf16_t)(-2.0f * z[i]);
}

// ---------------------------------------------------------------------------
// 2) codebook -> bf16 copy + e_sq[k] = sum_d e[k,d]^2  (f32)
// ---------------------------------------------------------------------------
__global__ void prep_cb_kernel(const float* __restrict__ cbf,
                               bf16_t* __restrict__ cb,
                               float*  __restrict__ e_sq) {
    __shared__ float sm[256];
    int k = blockIdx.x;
    int d = threadIdx.x;
    float v = cbf[k * Dc + d];
    cb[k * Dc + d] = (bf16_t)v;
    sm[d] = v * v;
    __syncthreads();
    #pragma unroll
    for (int s = 128; s > 0; s >>= 1) {
        if (d < s) sm[d] += sm[d + s];
        __syncthreads();
    }
    if (d == 0) e_sq[k] = sm[0];
}

// ---------------------------------------------------------------------------
// TDM: DMA one 64-row codebook tile (64 x 512B) into LDS with a 16B row pad
// (528B pitch). D# per ISA 8.3/8.4: group0 {count=1, lds_addr, gaddr, type=2},
// group1 {data_size=2B, pad_enable, pad_interval=128dw, pad_amount=4dw,
//         tensor_dim0=256, tensor_dim1=8192, tile_dim0=256, tile_dim1=64,
//         tensor_dim0_stride=256}, groups 2/3 zero (2-D tensor).
// This toolchain exposes the 6-arg builtin (clang-23 / therock-10.0 form):
//   (u32x4 g0, i32x8 g1, i32x4 g2, i32x4 g3, i32x8 pad, i32 cpol)
// ---------------------------------------------------------------------------
__device__ __forceinline__ void tdm_load_tile(const bf16_t* gptr, unsigned lds_off) {
    unsigned long long ga = (unsigned long long)(const void*)gptr;
    u32x4 g0;
    g0[0] = 1u;                                               // count=1 (valid)
    g0[1] = lds_off;                                          // lds_addr
    g0[2] = (unsigned)(ga & 0xffffffffull);                   // global_addr[31:0]
    g0[3] = (unsigned)((ga >> 32) & 0x01ffffffull)            // global_addr[56:32]
          | (2u << 30);                                       // type=2 (image)
    i32x8 g1;
    g1[0] = (1 << 16)            // data_size = 2 bytes
          | (1 << 20)            // pad_enable
          | (6 << 22)            // pad_interval: 2^(6+1)=128 DWORDs (512B)
          | (3 << 25);           // pad_amount: 3+1 = 4 DWORDs (16B)
    g1[1] = (int)(256u << 16);   // tensor_dim0[15:0]=256 elems
    g1[2] = (int)(8192u << 16);  // tensor_dim0 hi=0 | tensor_dim1 lo=8192 rows
    g1[3] = (int)(256u << 16);   // tensor_dim1 hi=0 | tile_dim0=256 elems
    g1[4] = TILE_N;              // tile_dim1=64 rows, tile_dim2=0
    g1[5] = 256;                 // tensor_dim0_stride[31:0]=256 elems
    g1[6] = 0;                   // stride hi / dim1_stride (unused, 2-D)
    g1[7] = 0;
    i32x4 gz4 = (i32x4){0, 0, 0, 0};
    i32x8 gz8 = (i32x8){0, 0, 0, 0, 0, 0, 0, 0};
    __builtin_amdgcn_tensor_load_to_lds(g0, g1, gz4, gz4, gz8, 0);
}

// ---------------------------------------------------------------------------
// 3) distance GEMM + argmin.
//    block = 256 threads = 8 waves; wave w owns M-tile (blockIdx*8 + w)*16
//    with the 16x256 bf16 A tile resident in registers (8 v16bf frags).
//    Codebook tiles (64 codes) are TDM-DMA'd into a double-buffered LDS
//    staging area (s_wait_tensorcnt / barrier pipelined so the DMA of tile
//    t+1 overlaps the 32 WMMAs of tile t). Per dc-group, all 4 B fragments
//    are loaded into distinct registers first (8x ds_load_b128 clause, one
//    dscnt wait), then 4 independent-chain WMMAs consume them while the next
//    group's loads issue.
// ---------------------------------------------------------------------------
__global__ void vq_argmin_kernel(const bf16_t* __restrict__ zb,
                                 const bf16_t* __restrict__ cb,
                                 const float*  __restrict__ e_sq,
                                 int*   __restrict__ idx_ws,
                                 float* __restrict__ out_idx) {
    __shared__ __align__(16) unsigned char btile[2][TILE_LDS];

    const int wave = threadIdx.x >> 5;
    const int lane = threadIdx.x & 31;
    const int h    = lane >> 4;        // which 16-lane half
    const int l16  = lane & 15;
    const int mbase = (blockIdx.x * 8 + wave) * 16;

    // low 32 bits of the flat address of an LDS object = its LDS byte offset
    const unsigned lds_base =
        (unsigned)(unsigned long long)(const void*)&btile[0][0];

    // kick off DMA of tile 0 while we assemble the A fragments
    if (wave == 0) tdm_load_tile(cb, lds_base);

    // ---- load resident A fragments (layout: ISA 7.12.2, 16-bit A 16x32) ----
    const bf16_t* zrow = zb + (mbase + l16) * Dc;
    v16bf a[8];
    #pragma unroll
    for (int dc = 0; dc < 8; ++dc) {
        const bf16_t* p = zrow + dc * 32 + h * 8;
        v8bf lo = *(const v8bf*)(p);
        v8bf hi = *(const v8bf*)(p + 16);
        a[dc] = __builtin_shufflevector(lo, hi,
                                        0, 1, 2, 3, 4, 5, 6, 7,
                                        8, 9, 10, 11, 12, 13, 14, 15);
    }

    float vmin[8];
    int   vidx[8];
    #pragma unroll
    for (int i = 0; i < 8; ++i) { vmin[i] = 3.4e38f; vidx[i] = 0; }

    for (int t = 0; t < NTILES; ++t) {
        const int nb  = t * TILE_N;
        const int cur = t & 1;

        if (wave == 0) {
            if (t + 1 < NTILES) {
                // queue next tile into the other buffer, then wait for tile t
                tdm_load_tile(cb + (size_t)(nb + TILE_N) * Dc,
                              lds_base + (unsigned)((cur ^ 1) * TILE_LDS));
                __builtin_amdgcn_s_wait_tensorcnt(1);
            } else {
                __builtin_amdgcn_s_wait_tensorcnt(0);
            }
        }
        __syncthreads();   // tile t visible to all waves

        // seed accumulators with ||e_n||^2 for the 4 sub-tiles of 16 codes
        v8f c[4];
        #pragma unroll
        for (int j = 0; j < 4; ++j) {
            const float es = e_sq[nb + j * 16 + l16];
            #pragma unroll
            for (int i = 0; i < 8; ++i) c[j][i] = es;
        }

        // B layout (32x16 bf16): lane = col N = l16 (codebook row j*16+l16),
        // K runs [dc*32 + h*8 .. +8) and [dc*32 + 16 + h*8 .. +8). Load all 4 fragments of this dc-group
        // into distinct registers, then run the 4 independent-chain WMMAs.
        const unsigned char* base = &btile[cur][0];
        #pragma unroll
        for (int dc = 0; dc < 8; ++dc) {
            v16bf b[4];
            #pragma unroll
            for (int j = 0; j < 4; ++j) {
                const bf16_t* p = (const bf16_t*)(base + (j * 16 + l16) * PITCH_B
                                                  + dc * 64 + h * 16);
                v8bf lo = *(const v8bf*)(p);
                v8bf hi = *(const v8bf*)(p + 16);
                b[j] = __builtin_shufflevector(lo, hi,
                                               0, 1, 2, 3, 4, 5, 6, 7,
                                               8, 9, 10, 11, 12, 13, 14, 15);
            }
            #pragma unroll
            for (int j = 0; j < 4; ++j) {
                c[j] = __builtin_amdgcn_wmma_f32_16x16x32_bf16(
                           false, a[dc], false, b[j], (short)0, c[j], false, false);
            }
        }

        // running argmin; ascending j keeps the lowest index on ties
        #pragma unroll
        for (int j = 0; j < 4; ++j) {
            const int n = nb + j * 16 + l16;
            #pragma unroll
            for (int i = 0; i < 8; ++i) {
                if (c[j][i] < vmin[i]) { vmin[i] = c[j][i]; vidx[i] = n; }
            }
        }

        __syncthreads();   // all waves done reading buf[cur] before it refills
    }

    // ---- cross-lane argmin within each 16-lane half (rows split by half) ----
    #pragma unroll
    for (int off = 1; off < 16; off <<= 1) {
        #pragma unroll
        for (int i = 0; i < 8; ++i) {
            float om = __shfl_xor(vmin[i], off, 16);
            int   oi = __shfl_xor(vidx[i], off, 16);
            bool take = (om < vmin[i]) || (om == vmin[i] && oi < vidx[i]);
            if (take) { vmin[i] = om; vidx[i] = oi; }
        }
    }

    // lanes 0-15 hold rows mbase+0..7, lanes 16-31 hold rows mbase+8..15
    if (l16 == 0) {
        const int rbase = mbase + h * 8;
        #pragma unroll
        for (int i = 0; i < 8; ++i) {
            idx_ws[rbase + i]  = vidx[i];
            out_idx[rbase + i] = (float)vidx[i];
        }
    }
}

// ---------------------------------------------------------------------------
// 4) gather q = codebook[idx], straight-through output, per-row loss partials
// ---------------------------------------------------------------------------
__global__ void gather_out_kernel(const float* __restrict__ z,
                                  const unsigned char* __restrict__ mask,
                                  const float* __restrict__ cbf,
                                  const int*   __restrict__ idx_ws,
                                  float* __restrict__ out_q,
                                  float* __restrict__ loss_part,
                                  float* __restrict__ mask_part) {
    __shared__ float sm[256];
    int r = blockIdx.x;
    int d = threadIdx.x;
    int idx = idx_ws[r];
    bool m = mask[r] != 0;
    float zz = z[r * Dc + d];
    float q  = cbf[idx * Dc + d];
    // quantized = z + sg(q - z) == q (forward); masked-out rows pass z
    out_q[r * Dc + d] = m ? q : zz;
    float diff = zz - q;
    sm[d] = m ? diff * diff : 0.0f;
    __syncthreads();
    #pragma unroll
    for (int s = 128; s > 0; s >>= 1) {
        if (d < s) sm[d] += sm[d + s];
        __syncthreads();
    }
    if (d == 0) {
        loss_part[r] = sm[0];
        mask_part[r] = m ? 1.0f : 0.0f;
    }
}

// ---------------------------------------------------------------------------
// 5) final fixed-order reduction -> commit loss scalar
// ---------------------------------------------------------------------------
__global__ void finalize_kernel(const float* __restrict__ loss_part,
                                const float* __restrict__ mask_part,
                                float* __restrict__ out_loss) {
    __shared__ float sl[256];
    __shared__ float sc[256];
    int t = threadIdx.x;
    float a = 0.0f, c = 0.0f;
    for (int i = t; i < Mc; i += 256) { a += loss_part[i]; c += mask_part[i]; }
    sl[t] = a; sc[t] = c;
    __syncthreads();
    #pragma unroll
    for (int s = 128; s > 0; s >>= 1) {
        if (t < s) { sl[t] += sl[t + s]; sc[t] += sc[t + s]; }
        __syncthreads();
    }
    if (t == 0) {
        float denom = fmaxf(sc[0] * (float)Dc, 1.0f);
        out_loss[0] = COMMIT_W * sl[0] / denom;
    }
}

// ---------------------------------------------------------------------------
extern "C" void kernel_launch(void* const* d_in, const int* in_sizes, int n_in,
                              void* d_out, int out_size, void* d_ws, size_t ws_size,
                              hipStream_t stream) {
    const float*         z    = (const float*)d_in[0];          // [B,T,D] f32
    const unsigned char* mask = (const unsigned char*)d_in[1];  // [B,T] bool
    const float*         cbf  = (const float*)d_in[2];          // [K,D] f32

    float* out = (float*)d_out;
    float* out_q    = out;                  // [Mc*Dc]
    float* out_idx  = out + Mc * Dc;        // [Mc] indices as float
    float* out_loss = out + Mc * Dc + Mc;   // [1]

    // workspace carve-up (all naturally aligned; ~12.8 MB total)
    char* ws = (char*)d_ws;
    bf16_t* zb        = (bf16_t*)ws;                 ws += (size_t)Mc * Dc * sizeof(bf16_t); // 8 MB
    bf16_t* cb        = (bf16_t*)ws;                 ws += (size_t)Kc * Dc * sizeof(bf16_t); // 4 MB
    float*  e_sq      = (float*)ws;                  ws += (size_t)Kc * sizeof(float);       // 32 KB
    int*    idx_ws    = (int*)ws;                    ws += (size_t)Mc * sizeof(int);         // 64 KB
    float*  loss_part = (float*)ws;                  ws += (size_t)Mc * sizeof(float);       // 64 KB
    float*  mask_part = (float*)ws;                  ws += (size_t)Mc * sizeof(float);       // 64 KB
    (void)ws_size; (void)in_sizes; (void)n_in; (void)out_size;

    prep_z_kernel   <<<(Mc * Dc) / 256, 256, 0, stream>>>(z, zb);
    prep_cb_kernel  <<<Kc, 256, 0, stream>>>(cbf, cb, e_sq);
    vq_argmin_kernel<<<Mc / 128, 256, 0, stream>>>(zb, cb, e_sq, idx_ws, out_idx);
    gather_out_kernel<<<Mc, 256, 0, stream>>>(z, mask, cbf, idx_ws, out_q,
                                              loss_part, mask_part);
    finalize_kernel <<<1, 256, 0, stream>>>(loss_part, mask_part, out_loss);
}
